// SIGReg_35055523070528
// MI455X (gfx1250) — matrix operand order repaired and checked by reading, not compiled, not run on hardware
//
#include <hip/hip_runtime.h>
#include <hip/hip_bf16.h>

// ---- problem constants (match reference) ----
#define DIM    1024      // embed dim (K of GEMM)
#define BATCH  8192      // columns (N of GEMM)
#define KP     4096      // rows (M of GEMM) = 32*128
#define EPS    1e-6f

// ---- tiling ----
#define STRIPS       16                      // column strips
#define STRIP_COLS   (BATCH / STRIPS)        // 512
#define GROUPS       (STRIP_COLS / 64)       // 8 column groups of 64
#define CHUNKS       (GROUPS * 32)           // k-chunks per strip

typedef __attribute__((ext_vector_type(16))) __bf16    v16bf;
typedef __attribute__((ext_vector_type(8)))  float     v8f;
typedef __attribute__((ext_vector_type(8)))  unsigned  v8u;
typedef __attribute__((ext_vector_type(4)))  int       v4i;

#define AS1 __attribute__((address_space(1)))
#define AS3 __attribute__((address_space(3)))

#if __has_builtin(__builtin_amdgcn_global_load_async_to_lds_b128)
#define HAVE_ASYNC_LDS 1
#else
#define HAVE_ASYNC_LDS 0
#endif

__device__ __forceinline__ unsigned short f32_to_bf16_bits(float x) {
    unsigned u = __float_as_uint(x);
    u += 0x7FFFu + ((u >> 16) & 1u);    // round-to-nearest-even
    return (unsigned short)(u >> 16);
}
__device__ __forceinline__ float bf16_bits_to_f32(unsigned short h) {
    return __uint_as_float(((unsigned)h) << 16);
}

__device__ __forceinline__ v8f wmma_bf16(v8u a, v8u b, v8f c) {
    return __builtin_amdgcn_wmma_f32_16x16x32_bf16(
        /*neg_a=*/false, __builtin_bit_cast(v16bf, a),
        /*neg_b=*/false, __builtin_bit_cast(v16bf, b),
        /*c_mod=*/(short)0, c, /*reuse_a=*/false, /*reuse_b=*/false);
}

__device__ __forceinline__ v8u assemble8(uint4 q0, uint4 q1) {
    v8u r;
    r[0] = q0.x; r[1] = q0.y; r[2] = q0.z; r[3] = q0.w;
    r[4] = q1.x; r[5] = q1.y; r[6] = q1.z; r[7] = q1.w;
    return r;
}

// Wait for this wave's outstanding async LDS-DMA loads.
__device__ __forceinline__ void wait_async() {
#if HAVE_ASYNC_LDS
#if __has_builtin(__builtin_amdgcn_s_wait_asynccnt)
    __builtin_amdgcn_s_wait_asynccnt(0);
#else
    asm volatile("s_wait_asynccnt 0" ::: "memory");
#endif
#endif
}

// Stage one 64x32 bf16 B tile chunk (hi+lo) into LDS: 16 B per thread per array.
__device__ __forceinline__ void stage_chunk(
    const unsigned short* __restrict__ embHi,
    const unsigned short* __restrict__ embLo,
    unsigned short* dstHi, unsigned short* dstLo,
    int tid, int colBase, int k)
{
    int idx = tid * 8;                   // 0..2040, multiple of 8 (16 B aligned)
    int col = idx >> 5, kk = idx & 31;
    size_t goff = (size_t)(colBase + col) * DIM + k + kk;
#if HAVE_ASYNC_LDS
    // gfx1250 LDS-DMA: no VGPR round trip, tracked by ASYNCcnt.
    __builtin_amdgcn_global_load_async_to_lds_b128(
        (AS1 v4i*)(AS1 void*)(embHi + goff),
        (AS3 v4i*)(AS3 void*)(dstHi + idx), 0, 0);
    __builtin_amdgcn_global_load_async_to_lds_b128(
        (AS1 v4i*)(AS1 void*)(embLo + goff),
        (AS3 v4i*)(AS3 void*)(dstLo + idx), 0, 0);
#else
    uint4 vh = *(const uint4*)(embHi + goff);
    uint4 vl = *(const uint4*)(embLo + goff);
    *(uint4*)(dstHi + idx) = vh;
    *(uint4*)(dstLo + idx) = vl;
#endif
}

__global__ void zero_kernel(float* p, int n) {
    int i = blockIdx.x * blockDim.x + threadIdx.x;
    if (i < n) p[i] = 0.f;
}

// One-pass f32 -> (bf16 hi, bf16 lo) split. Memory bound; runs once.
__global__ __launch_bounds__(256) void cvt_split_kernel(
    const float* __restrict__ in,
    unsigned short* __restrict__ hi,
    unsigned short* __restrict__ lo, int n4)
{
    int i = blockIdx.x * blockDim.x + threadIdx.x;
    if (i >= n4) return;
    float4 v = ((const float4*)in)[i];
    unsigned short h0 = f32_to_bf16_bits(v.x), h1 = f32_to_bf16_bits(v.y);
    unsigned short h2 = f32_to_bf16_bits(v.z), h3 = f32_to_bf16_bits(v.w);
    unsigned short l0 = f32_to_bf16_bits(v.x - bf16_bits_to_f32(h0));
    unsigned short l1 = f32_to_bf16_bits(v.y - bf16_bits_to_f32(h1));
    unsigned short l2 = f32_to_bf16_bits(v.z - bf16_bits_to_f32(h2));
    unsigned short l3 = f32_to_bf16_bits(v.w - bf16_bits_to_f32(h3));
    uint2 hq = make_uint2((unsigned)h0 | ((unsigned)h1 << 16),
                          (unsigned)h2 | ((unsigned)h3 << 16));
    uint2 lq = make_uint2((unsigned)l0 | ((unsigned)l1 << 16),
                          (unsigned)l2 | ((unsigned)l3 << 16));
    ((uint2*)hi)[i] = hq;
    ((uint2*)lo)[i] = lq;
}

// Fused GEMM + power-sum kernel, bf16 split inputs, double-buffered async B staging.
// grid = (STRIPS col-strips of STRIP_COLS, 32 row-blocks of 128), block = 256 (8 waves).
__global__ __launch_bounds__(256) void moments_kernel(
    const unsigned short* __restrict__ embHi,   // [BATCH][DIM] bf16 bits
    const unsigned short* __restrict__ embLo,
    const unsigned short* __restrict__ projHi,  // [KP][DIM] bf16 bits
    const unsigned short* __restrict__ projLo,
    float* __restrict__ mom)                    // [KP][4]
{
    __shared__ __align__(16) unsigned short BtHi[2][64 * 32];  // 2 x 4 KB
    __shared__ __align__(16) unsigned short BtLo[2][64 * 32];

    const int tid     = threadIdx.x;
    const int wave    = tid >> 5;
    const int lane    = tid & 31;
    const int half    = lane >> 4;       // 0: lanes 0-15, 1: lanes 16-31
    const int laneCol = lane & 15;
    const int rowBase   = blockIdx.y * 128 + wave * 16;
    const int stripBase = blockIdx.x * STRIP_COLS;

    float s1[8], s2[8], s3[8], s4[8];
#pragma unroll
    for (int i = 0; i < 8; ++i) { s1[i] = 0.f; s2[i] = 0.f; s3[i] = 0.f; s4[i] = 0.f; }

    // A-matrix lane row: M = laneCol (both lane halves, per ISA 16-bit A layout).
    const uint4* pAhi = (const uint4*)(projHi + (size_t)(rowBase + laneCol) * DIM);
    const uint4* pAlo = (const uint4*)(projLo + (size_t)(rowBase + laneCol) * DIM);

    // Prologue: stage first chunk into buffer 0.
    int cur = 0;
    stage_chunk(embHi, embLo, &BtHi[0][0], &BtLo[0][0], tid, stripBase, 0);

    for (int g = 0; g < GROUPS; ++g) {
        v8f acc[4] = {{0.f,0.f,0.f,0.f,0.f,0.f,0.f,0.f},
                      {0.f,0.f,0.f,0.f,0.f,0.f,0.f,0.f},
                      {0.f,0.f,0.f,0.f,0.f,0.f,0.f,0.f},
                      {0.f,0.f,0.f,0.f,0.f,0.f,0.f,0.f}};

        for (int ki = 0; ki < 32; ++ki) {
            // Current chunk's DMA complete (own ops), then all waves' chunks visible.
            wait_async();
            __syncthreads();

            // Kick off next chunk's DMA into the other buffer; overlaps the WMMAs.
            int nc = g * 32 + ki + 1;
            if (nc < CHUNKS) {
                int ng = nc >> 5, nki = nc & 31;
                stage_chunk(embHi, embLo, &BtHi[cur ^ 1][0], &BtLo[cur ^ 1][0],
                            tid, stripBase + ng * 64, nki * 32);
            }

            const int k = ki * 32;
            // ---- A chunk (ISA 16-bit A 16x32 layout):
            // lanes<16 -> K {0-7,16-23}, lanes>=16 -> K {8-15,24-31}; 8 bf16 = one uint4.
            const int i0 = (k + half * 8) >> 3;
            const int i1 = (k + 16 + half * 8) >> 3;
            v8u a_hi = assemble8(pAhi[i0], pAhi[i1]);
            v8u a_lo = assemble8(pAlo[i0], pAlo[i1]);

            // ---- 4 column tiles share the A registers
#pragma unroll
            for (int t = 0; t < 4; ++t) {
                // B layout (32x16 bf16): lane col N=laneCol; half 0 -> K {0-7,16-23}, half 1 -> K {8-15,24-31}
                int boff = (t * 16 + laneCol) * 32 + half * 8;    // 16 B aligned
                const uint4* pbh = (const uint4*)&BtHi[cur][boff];
                const uint4* pbl = (const uint4*)&BtLo[cur][boff];
                v8u b_hi = assemble8(pbh[0], pbh[2]);
                v8u b_lo = assemble8(pbl[0], pbl[2]);

                // f32 via bf16 split: hi*hi + hi*lo + lo*hi (drop lo*lo, ~2^-16 rel err)
                acc[t] = wmma_bf16(a_hi, b_hi, acc[t]);
                acc[t] = wmma_bf16(a_hi, b_lo, acc[t]);
                acc[t] = wmma_bf16(a_lo, b_hi, acc[t]);
            }
            cur ^= 1;
        }

        // Fold this 16x64 C tile into per-row power sums.
        // C layout: c[i] = C[row = i + 8*half][col = laneCol] per tile.
#pragma unroll
        for (int t = 0; t < 4; ++t) {
#pragma unroll
            for (int i = 0; i < 8; ++i) {
                float x  = acc[t][i];
                float x2 = x * x;
                s1[i] += x;
                s2[i] += x2;
                s3[i] += x2 * x;
                s4[i] += x2 * x2;
            }
        }
    }

    // Butterfly-reduce across the 16 column lanes of each half (masks stay in-half).
#pragma unroll
    for (int i = 0; i < 8; ++i) {
#pragma unroll
        for (int m = 8; m >= 1; m >>= 1) {
            s1[i] += __shfl_xor(s1[i], m, 32);
            s2[i] += __shfl_xor(s2[i], m, 32);
            s3[i] += __shfl_xor(s3[i], m, 32);
            s4[i] += __shfl_xor(s4[i], m, 32);
        }
    }

    // Lanes 0..7 of each half commit one row each.
    if (laneCol < 8) {
        int j = laneCol;
        int row = rowBase + half * 8 + j;
        atomicAdd(&mom[row * 4 + 0], s1[j]);
        atomicAdd(&mom[row * 4 + 1], s2[j]);
        atomicAdd(&mom[row * 4 + 2], s3[j]);
        atomicAdd(&mom[row * 4 + 3], s4[j]);
    }
}

// Raw moments -> per-row loss -> scalar mean. Single block.
__global__ __launch_bounds__(256) void loss_kernel(const float* __restrict__ mom,
                                                   float* __restrict__ out)
{
    __shared__ float red[256];
    const int tid = threadIdx.x;
    const float invN = 1.0f / (float)BATCH;
    float local = 0.f;
    for (int r = tid; r < KP; r += 256) {
        float mu  = mom[r * 4 + 0] * invN;
        float ex2 = mom[r * 4 + 1] * invN;
        float ex3 = mom[r * 4 + 2] * invN;
        float ex4 = mom[r * 4 + 3] * invN;
        float var = ex2 - mu * mu;
        var = var > 0.f ? var : 0.f;
        float sd  = sqrtf(var) + EPS;
        float mu2 = mu * mu;
        float c3  = ex3 - 3.f * mu * ex2 + 2.f * mu * mu2;
        float c4  = ex4 - 4.f * mu * ex3 + 6.f * mu2 * ex2 - 3.f * mu2 * mu2;
        float is  = 1.f / sd;
        float is2 = is * is;
        float m3  = c3 * is2 * is;
        float m4  = c4 * is2 * is2;
        float a   = sd - 1.f;
        float b   = m4 - 3.f;
        local += mu2 + a * a + 0.1f * m3 * m3 + 0.1f * b * b;
    }
    red[tid] = local;
    __syncthreads();
    for (int s = 128; s > 0; s >>= 1) {
        if (tid < s) red[tid] += red[tid + s];
        __syncthreads();
    }
    if (tid == 0) out[0] = red[0] * (1.0f / (float)KP);
}

extern "C" void kernel_launch(void* const* d_in, const int* in_sizes, int n_in,
                              void* d_out, int out_size, void* d_ws, size_t ws_size,
                              hipStream_t stream) {
    const float* emb  = (const float*)d_in[0];   // [8192][1024] f32
    const float* proj = (const float*)d_in[1];   // [32*128][1024] f32
    float* out = (float*)d_out;                  // scalar f32

    // ---- workspace layout (bytes) ----
    // mom:    [0, 64K)                         4096*4 f32
    // embHi:  [64K, 64K+16M)                   8192*1024 bf16 bits
    // embLo:  next 16M
    // projHi: next 8M
    // projLo: next 8M                          total ~48.1 MB
    char* ws = (char*)d_ws;
    float* mom = (float*)ws;
    const size_t MOMB   = (size_t)KP * 4 * sizeof(float);          // 64 KB
    const size_t EMB_N  = (size_t)BATCH * DIM;                     // 8.4M
    const size_t PROJ_N = (size_t)KP * DIM;                        // 4.2M
    unsigned short* embHi  = (unsigned short*)(ws + MOMB);
    unsigned short* embLo  = embHi  + EMB_N;
    unsigned short* projHi = embLo  + EMB_N;
    unsigned short* projLo = projHi + PROJ_N;

    const int momN = KP * 4;
    zero_kernel<<<(momN + 255) / 256, 256, 0, stream>>>(mom, momN);

    const int embN4  = (int)(EMB_N / 4);
    const int projN4 = (int)(PROJ_N / 4);
    cvt_split_kernel<<<(embN4  + 255) / 256, 256, 0, stream>>>(emb,  embHi,  embLo,  embN4);
    cvt_split_kernel<<<(projN4 + 255) / 256, 256, 0, stream>>>(proj, projHi, projLo, projN4);

    moments_kernel<<<dim3(STRIPS, 32), 256, 0, stream>>>(embHi, embLo, projHi, projLo, mom);
    loss_kernel<<<1, 256, 0, stream>>>(mom, out);
}
